// EfficientMHA_76759655514547
// MI455X (gfx1250) — matrix operand order repaired and checked by reading, not compiled, not run on hardware
//
#include <hip/hip_runtime.h>

// MHA forward: B=4, L=2048, D=1024, H=16, DH=64 (f32 in/out).
// All matmuls via v_wmma_f32_16x16x32_bf16; bf16 tiles staged through LDS,
// with CDNA5 GLOBAL_LOAD_ASYNC_TO_LDS (ASYNCcnt) double-buffering in the
// attention kernel and async A-staging in the output GEMM.
//  K1: qkv = x@Wqkv + b -> Q,K bf16 [B,H,L,64] (Q pre-scaled 1/8), V bf16 [B,H,64,L]
//  K2: flash attention  -> ctx bf16 [B,L,D], validf f32 [B,L]
//  K3: out = (ctx@Wout + bout) * validf

#define B_  4
#define L_  2048
#define D_  1024
#define H_  16
#define DH_ 64
#define N3_ 3072
#define NKB (L_ / 64)

typedef __attribute__((ext_vector_type(16))) __bf16 v16bf;
typedef __attribute__((ext_vector_type(8)))  float  v8f;
typedef unsigned short u16;

struct alignas(16) H8 { u16 h[8]; };           // 8 bf16 = 16B -> b128 ops
union  AFragU { v16bf v; H8 h8[2]; };

__device__ __forceinline__ u16 f2bf(float f) { // RNE f32->bf16
  unsigned u = __float_as_uint(f);
  u += 0x7FFFu + ((u >> 16) & 1u);
  return (u16)(u >> 16);
}

__device__ __forceinline__ v8f wmma_bf16(v16bf a, v16bf b, v8f c) {
  return __builtin_amdgcn_wmma_f32_16x16x32_bf16(false, a, false, b, (short)0, c,
                                                 false, false);
}

// CDNA5 async DMA: global -> LDS, 16B per lane, tracked by ASYNCcnt.
__device__ __forceinline__ void async_b128(unsigned lds_off, const void* g) {
  unsigned long long ga = (unsigned long long)(uintptr_t)g;
  asm volatile("global_load_async_to_lds_b128 %0, %1, off"
               :: "v"(lds_off), "v"(ga) : "memory");
}
__device__ __forceinline__ unsigned lds_lo32(const void* p) {
  return (unsigned)(uintptr_t)p;   // low 32 bits of generic shared ptr = LDS offset
}

// ---------------------------------------------------------------------------
// K1: C[8192,3072] = x @ Wqkv + bqkv. Block 256 thr (8 waves), tile 128x128,
// K-step 32; 8 WMMAs per barrier pair per wave. Scatter to Q,K ([B,H,L,64])
// and V ([B,H,64,L], packed 16B stores).
// ---------------------------------------------------------------------------
__global__ __launch_bounds__(256) void qkv_gemm(
    const float* __restrict__ x, const float* __restrict__ W,
    const float* __restrict__ bias, u16* __restrict__ Qo, u16* __restrict__ Ko,
    u16* __restrict__ Vo) {
  __shared__ u16 As[128 * 40];   // 128x32 bf16, row stride 80B (16B aligned)
  __shared__ u16 Bs[128 * 40];   // 32x128 stored transposed [n][k]

  const int tid  = threadIdx.x;
  const int bid  = blockIdx.x;
  const int m0   = (bid / (N3_ / 128)) * 128;
  const int n0   = (bid % (N3_ / 128)) * 128;
  const int wave = tid >> 5, lane = tid & 31;
  const int wm   = wave & 3, wn = wave >> 2;   // 4 x 2 wave grid: 32x64 per wave
  const int arow = lane & 15, kg = lane >> 4;

  v8f acc[2][4] = {};

  for (int k0 = 0; k0 < D_; k0 += 32) {
#pragma unroll
    for (int i = 0; i < 4; ++i) {              // A: 128x32 f32 -> bf16
      int idx = tid + i * 256;
      int row = idx >> 3, c4 = idx & 7;
      float4 f = *(const float4*)(x + (size_t)(m0 + row) * D_ + k0 + c4 * 4);
      u16* d = &As[row * 40 + c4 * 4];
      d[0] = f2bf(f.x); d[1] = f2bf(f.y); d[2] = f2bf(f.z); d[3] = f2bf(f.w);
    }
#pragma unroll
    for (int i = 0; i < 4; ++i) {              // B: 32x128 f32, transpose to [n][k]
      int idx = tid + i * 256;
      int kk = idx >> 5, n4 = idx & 31;
      float4 f = *(const float4*)(W + (size_t)(k0 + kk) * N3_ + n0 + n4 * 4);
      Bs[(n4 * 4 + 0) * 40 + kk] = f2bf(f.x);
      Bs[(n4 * 4 + 1) * 40 + kk] = f2bf(f.y);
      Bs[(n4 * 4 + 2) * 40 + kk] = f2bf(f.z);
      Bs[(n4 * 4 + 3) * 40 + kk] = f2bf(f.w);
    }
    __syncthreads();

    AFragU a[2], b[4];
#pragma unroll
    for (int i = 0; i < 2; ++i) {
      const u16* p = &As[(wm * 32 + i * 16 + arow) * 40];
      a[i].h8[0] = *(const H8*)&p[kg * 16];
      a[i].h8[1] = *(const H8*)&p[kg * 16 + 8];
    }
#pragma unroll
    for (int j = 0; j < 4; ++j) {
      const u16* p = &Bs[(wn * 64 + j * 16 + arow) * 40 + kg * 16];
      b[j].h8[0] = *(const H8*)&p[0];
      b[j].h8[1] = *(const H8*)&p[8];
    }
#pragma unroll
    for (int i = 0; i < 2; ++i)
#pragma unroll
      for (int j = 0; j < 4; ++j) acc[i][j] = wmma_bf16(a[i].v, b[j].v, acc[i][j]);
    __syncthreads();
  }

  const int half = lane >> 4, cl = lane & 15;
#pragma unroll
  for (int i = 0; i < 2; ++i) {
#pragma unroll
    for (int j = 0; j < 4; ++j) {
      int col   = n0 + wn * 64 + j * 16 + cl;
      int which = col >> 10;                   // 0=Q 1=K 2=V
      int dd    = col & 1023;
      int h     = dd >> 6, dh = dd & 63;
      float bv  = bias[col];
      float sc  = (which == 0) ? 0.125f : 1.0f;
      int rbase = m0 + wm * 32 + i * 16 + half * 8;   // 8 consecutive rows
      int bb = rbase >> 11, l0 = rbase & (L_ - 1);
      if (which == 2) {                        // V: [B,H,DH,L]; pack 8 rows -> 16B
        H8 pk;
#pragma unroll
        for (int r = 0; r < 8; ++r) pk.h[r] = f2bf(acc[i][j][r] + bv);
        *(H8*)&Vo[((size_t)(bb * H_ + h) * DH_ + dh) * L_ + l0] = pk;
      } else {                                 // Q/K: [B,H,L,DH]
        u16* dst = (which == 0) ? Qo : Ko;
#pragma unroll
        for (int r = 0; r < 8; ++r)
          dst[((size_t)(bb * H_ + h) * L_ + l0 + r) * DH_ + dh] =
              f2bf((acc[i][j][r] + bv) * sc);
      }
    }
  }
}

// ---------------------------------------------------------------------------
// K2: flash attention, 128 thr (4 waves), 64-query tile per block.
// K/V tiles double-buffered via GLOBAL_LOAD_ASYNC_TO_LDS_B128 (prefetch kb+1
// while computing kb). Mask fast path: block-uniform all-ones check.
// ---------------------------------------------------------------------------
__global__ __launch_bounds__(128) void attn_kernel(
    const u16* __restrict__ Q, const u16* __restrict__ K,
    const u16* __restrict__ V, const unsigned char* __restrict__ mask,
    u16* __restrict__ ctx, float* __restrict__ validf) {
  __shared__ u16 Qs[64 * 72];          // [qrow][d], stride 144B
  __shared__ u16 Ks[2][64 * 72];       // [key][d]
  __shared__ u16 Vs[2][64 * 72];       // [d][key] (V already d-major in ws)
  __shared__ u16 Ps[4][16 * 72];       // per-wave P strip

  const int tid  = threadIdx.x;
  const int bid  = blockIdx.x;
  const int qt   = bid & 31;
  const int h    = (bid >> 5) & 15;
  const int b    = bid >> 9;
  const int q0   = qt * 64;
  const int wave = tid >> 5, lane = tid & 31;
  const int arow = lane & 15, kg = lane >> 4;
  const int half = kg, cl = arow;

  const u16* Qg = Q + (size_t)(b * H_ + h) * L_ * DH_;
  const u16* Kg = K + (size_t)(b * H_ + h) * L_ * DH_;
  const u16* Vg = V + (size_t)(b * H_ + h) * DH_ * L_;

  auto issue = [&](int kb, int buf) {          // 8 async b128 per thread
    const int kbase = kb * 64;
#pragma unroll
    for (int i = 0; i < 4; ++i) {
      int idx = tid + i * 128;
      int row = idx >> 3, c8 = idx & 7;
      async_b128(lds_lo32(&Ks[buf][row * 72 + c8 * 8]),
                 Kg + (size_t)(kbase + row) * DH_ + c8 * 8);
      async_b128(lds_lo32(&Vs[buf][row * 72 + c8 * 8]),
                 Vg + (size_t)row * L_ + kbase + c8 * 8);
    }
  };

  issue(0, 0);                                 // start DMA before Q staging

#pragma unroll
  for (int i = 0; i < 4; ++i) {                // stage Q tile (plain copies)
    int idx = tid + i * 128;
    int row = idx >> 3, c8 = idx & 7;
    *(H8*)&Qs[row * 72 + c8 * 8] =
        *(const H8*)&Qg[(size_t)(q0 + row) * DH_ + c8 * 8];
  }

  float mrow[8], lrow[8];
  unsigned anyv[8];
  v8f o[4] = {};
#pragma unroll
  for (int r = 0; r < 8; ++r) { mrow[r] = -3.0e38f; lrow[r] = 0.f; anyv[r] = 0u; }

  for (int kb = 0; kb < NKB; ++kb) {
    const int cur = kb & 1;
    const int kbase = kb * 64;
    if (kb + 1 < NKB) issue(kb + 1, cur ^ 1);  // prefetch next block

    int pred = 1;                              // mask-tile all-ones probe (32B/thread)
#pragma unroll
    for (int i = 0; i < 2; ++i) {
      int idx = tid + i * 128;
      int row = idx >> 2, c16 = idx & 3;
      uint4 mq = *(const uint4*)(mask + ((size_t)b * L_ + q0 + row) * L_ +
                                 kbase + c16 * 16);
      pred &= (mq.x == 0x01010101u) & (mq.y == 0x01010101u) &
              (mq.z == 0x01010101u) & (mq.w == 0x01010101u);
    }
    if (kb + 1 < NKB) asm volatile("s_wait_asynccnt 0x8" ::: "memory");
    else              asm volatile("s_wait_asynccnt 0x0" ::: "memory");
    const int fast = __syncthreads_and(pred);  // barrier: cur buffer now visible

    v8f s[4] = {};                             // S = Q K^T (16x64 per wave)
#pragma unroll
    for (int kc = 0; kc < 2; ++kc) {
      AFragU a;
      const u16* qp = &Qs[(wave * 16 + arow) * 72 + kc * 32];
      a.h8[0] = *(const H8*)&qp[kg * 16];
      a.h8[1] = *(const H8*)&qp[kg * 16 + 8];
#pragma unroll
      for (int t = 0; t < 4; ++t) {
        AFragU bb;
        const u16* kp = &Ks[cur][(t * 16 + cl) * 72 + kc * 32 + kg * 16];
        bb.h8[0] = *(const H8*)&kp[0];
        bb.h8[1] = *(const H8*)&kp[8];
        s[t] = wmma_bf16(a.v, bb.v, s[t]);
      }
    }

    if (fast) {                                // mask all true: clamp only
#pragma unroll
      for (int t = 0; t < 4; ++t)
#pragma unroll
        for (int r = 0; r < 8; ++r)
          s[t][r] = fminf(fmaxf(s[t][r], -10000.f), 10000.f);
#pragma unroll
      for (int r = 0; r < 8; ++r) anyv[r] = 1u;
    } else {                                   // rare: per-element mask
#pragma unroll
      for (int t = 0; t < 4; ++t) {
#pragma unroll
        for (int r = 0; r < 8; ++r) {
          int grow = q0 + wave * 16 + r + half * 8;
          int gcol = kbase + t * 16 + cl;
          unsigned char mv = mask[((size_t)b * L_ + grow) * L_ + gcol];
          float sv = fminf(fmaxf(s[t][r], -10000.f), 10000.f);
          if (!mv) sv = -10000.f;
          anyv[r] |= (unsigned)mv;
          s[t][r] = sv;
        }
      }
    }

#pragma unroll
    for (int r = 0; r < 8; ++r) {              // online softmax per row
      float mx = fmaxf(fmaxf(s[0][r], s[1][r]), fmaxf(s[2][r], s[3][r]));
#pragma unroll
      for (int d = 1; d < 16; d <<= 1) mx = fmaxf(mx, __shfl_xor(mx, d, 32));
      float mnew  = fmaxf(mrow[r], mx);
      float alpha = __expf(mrow[r] - mnew);
      mrow[r] = mnew;
      float psum = 0.f;
#pragma unroll
      for (int t = 0; t < 4; ++t) {
        float p = __expf(s[t][r] - mnew);
        s[t][r] = p;
        psum += p;
      }
#pragma unroll
      for (int d = 1; d < 16; d <<= 1) psum += __shfl_xor(psum, d, 32);
      lrow[r] = lrow[r] * alpha + psum;
#pragma unroll
      for (int t = 0; t < 4; ++t) o[t][r] *= alpha;
    }

#pragma unroll
    for (int t = 0; t < 4; ++t)                // P: C-layout -> A-layout via LDS
#pragma unroll
      for (int r = 0; r < 8; ++r)
        Ps[wave][(r + half * 8) * 72 + t * 16 + cl] = f2bf(s[t][r]);

#pragma unroll
    for (int kc = 0; kc < 2; ++kc) {           // O += P @ V
      AFragU a;
      const u16* pp = &Ps[wave][arow * 72 + kc * 32];
      a.h8[0] = *(const H8*)&pp[kg * 16];
      a.h8[1] = *(const H8*)&pp[kg * 16 + 8];
#pragma unroll
      for (int t = 0; t < 4; ++t) {
        AFragU bb;
        const u16* vp = &Vs[cur][(t * 16 + cl) * 72 + kc * 32 + kg * 16];
        bb.h8[0] = *(const H8*)&vp[0];
        bb.h8[1] = *(const H8*)&vp[8];
        o[t] = wmma_bf16(a.v, bb.v, o[t]);
      }
    }
    __syncthreads();                           // protect cur buf before re-issue
  }

#pragma unroll
  for (int r = 0; r < 8; ++r) {
    unsigned av = anyv[r];
#pragma unroll
    for (int d = 1; d < 16; d <<= 1) av |= (unsigned)__shfl_xor((int)av, d, 32);
    float inv = (av && lrow[r] > 0.f) ? 1.0f / lrow[r] : 0.f;
    int grow = q0 + wave * 16 + r + half * 8;
#pragma unroll
    for (int t = 0; t < 4; ++t)
      ctx[(size_t)(b * L_ + grow) * D_ + h * DH_ + t * 16 + cl] =
          f2bf(o[t][r] * inv);
    if (cl == 0) validf[b * L_ + grow] = av ? 1.f : 0.f;
  }
}

// ---------------------------------------------------------------------------
// K3: out[8192,1024] = (ctx @ Wout + bout) * validf[row].  ctx is bf16 ->
// A-tile staged by async-to-LDS; W converted f32->bf16 on the fly.
// ---------------------------------------------------------------------------
__global__ __launch_bounds__(256) void out_gemm(
    const u16* __restrict__ Cin, const float* __restrict__ W,
    const float* __restrict__ bias, const float* __restrict__ validf,
    float* __restrict__ out) {
  __shared__ u16 As[128 * 40];
  __shared__ u16 Bs[128 * 40];

  const int tid  = threadIdx.x;
  const int bid  = blockIdx.x;
  const int m0   = (bid / (D_ / 128)) * 128;
  const int n0   = (bid % (D_ / 128)) * 128;
  const int wave = tid >> 5, lane = tid & 31;
  const int wm   = wave & 3, wn = wave >> 2;
  const int arow = lane & 15, kg = lane >> 4;

  v8f acc[2][4] = {};

  for (int k0 = 0; k0 < D_; k0 += 32) {
#pragma unroll
    for (int i = 0; i < 2; ++i) {              // A: bf16 tile via async DMA
      int idx = tid + i * 256;
      int row = idx >> 2, c8 = idx & 3;
      async_b128(lds_lo32(&As[row * 40 + c8 * 8]),
                 Cin + (size_t)(m0 + row) * D_ + k0 + c8 * 8);
    }
#pragma unroll
    for (int i = 0; i < 4; ++i) {              // B: 32x128 f32 -> bf16 transposed
      int idx = tid + i * 256;
      int kk = idx >> 5, n4 = idx & 31;
      float4 f = *(const float4*)(W + (size_t)(k0 + kk) * D_ + n0 + n4 * 4);
      Bs[(n4 * 4 + 0) * 40 + kk] = f2bf(f.x);
      Bs[(n4 * 4 + 1) * 40 + kk] = f2bf(f.y);
      Bs[(n4 * 4 + 2) * 40 + kk] = f2bf(f.z);
      Bs[(n4 * 4 + 3) * 40 + kk] = f2bf(f.w);
    }
    asm volatile("s_wait_asynccnt 0x0" ::: "memory");
    __syncthreads();

    AFragU a[2], b[4];
#pragma unroll
    for (int i = 0; i < 2; ++i) {
      const u16* p = &As[(wm * 32 + i * 16 + arow) * 40];
      a[i].h8[0] = *(const H8*)&p[kg * 16];
      a[i].h8[1] = *(const H8*)&p[kg * 16 + 8];
    }
#pragma unroll
    for (int j = 0; j < 4; ++j) {
      const u16* p = &Bs[(wn * 64 + j * 16 + arow) * 40 + kg * 16];
      b[j].h8[0] = *(const H8*)&p[0];
      b[j].h8[1] = *(const H8*)&p[8];
    }
#pragma unroll
    for (int i = 0; i < 2; ++i)
#pragma unroll
      for (int j = 0; j < 4; ++j) acc[i][j] = wmma_bf16(a[i].v, b[j].v, acc[i][j]);
    __syncthreads();
  }

  const int half = lane >> 4, cl = lane & 15;
#pragma unroll
  for (int i = 0; i < 2; ++i) {
#pragma unroll
    for (int j = 0; j < 4; ++j) {
      int col  = n0 + wn * 64 + j * 16 + cl;
      float bv = bias[col];
#pragma unroll
      for (int r = 0; r < 8; ++r) {
        int row = m0 + wm * 32 + i * 16 + r + half * 8;
        out[(size_t)row * D_ + col] = (acc[i][j][r] + bv) * validf[row];
      }
    }
  }
}

// ---------------------------------------------------------------------------
extern "C" void kernel_launch(void* const* d_in, const int* in_sizes, int n_in,
                              void* d_out, int out_size, void* d_ws,
                              size_t ws_size, hipStream_t stream) {
  const float*         x    = (const float*)d_in[0];
  const unsigned char* mask = (const unsigned char*)d_in[1];
  const float*         Wqkv = (const float*)d_in[2];
  const float*         bqkv = (const float*)d_in[3];
  const float*         Wout = (const float*)d_in[4];
  const float*         bout = (const float*)d_in[5];
  float*               out  = (float*)d_out;

  // Workspace (~64.03 MB): Q | K | V bf16 (16 MB each), ctx bf16 (16 MB),
  // validf f32 (32 KB).
  char*  ws  = (char*)d_ws;
  const size_t QKV_ELEMS = (size_t)B_ * H_ * L_ * DH_;   // 8388608
  u16*   qb  = (u16*)ws;
  u16*   kb  = qb + QKV_ELEMS;
  u16*   vb  = kb + QKV_ELEMS;
  u16*   ctx = vb + QKV_ELEMS;
  float* valid = (float*)(ctx + (size_t)B_ * L_ * D_);

  qkv_gemm<<<dim3((8192 / 128) * (N3_ / 128)), dim3(256), 0, stream>>>(
      x, Wqkv, bqkv, qb, kb, vb);
  attn_kernel<<<dim3(B_ * H_ * (L_ / 64)), dim3(128), 0, stream>>>(
      qb, kb, vb, mask, ctx, valid);
  out_gemm<<<dim3((8192 / 128) * (D_ / 128)), dim3(256), 0, stream>>>(
      ctx, Wout, bout, valid, out);
}
